// EGNNPooling_46574625358253
// MI455X (gfx1250) — compile-verified, matrix operand order checked by reading, not run on hardware
//
#include <hip/hip_runtime.h>
#include <cstdint>

// ---------------- problem constants ----------------
#define CB 32
#define CN 4096
#define CE 65536
#define CD 32
#define CPAD 1
#define CKER 3
#define CSTR 2
#define CNPAD 4098            // N + 2*PAD
#define CP 2048               // pooled nodes per graph
#define CNTOT 6146            // NPAD + P
#define CPK 6144              // P*KERNEL
#define CEPG 77824            // E + 2*P*KERNEL
#define CNNODE (CB * CNTOT)   // 196672
#define CNEDGE ((long)CB * CEPG) // 2490368
#define HOUT ((size_t)CB * CP * CD) // h_pool elements in d_out

typedef __attribute__((ext_vector_type(16))) __bf16 v16bf;
typedef __attribute__((ext_vector_type(8)))  float  v8f;

#define WAVE_LDS_SYNC() asm volatile("s_wait_dscnt 0" ::: "memory")

// ---------------- workspace layout ----------------
static constexpr size_t AL(size_t x) { return (x + 255) & ~(size_t)255; }
static constexpr size_t OFF_HHBF = 0;
static constexpr size_t OFF_XX   = AL(OFF_HHBF + (size_t)CNNODE * CD * 2);
static constexpr size_t OFF_HN0  = AL(OFF_XX   + (size_t)CNNODE * 3 * 4);
static constexpr size_t OFF_HNBF = AL(OFF_HN0  + (size_t)CNNODE * CD * 4);
static constexpr size_t OFF_MAGG = AL(OFF_HNBF + (size_t)CNNODE * CD * 2);
static constexpr size_t OFF_XACC = AL(OFF_MAGG + (size_t)CNNODE * CD * 4);
static constexpr size_t OFF_CNT  = AL(OFF_XACC + (size_t)CNNODE * 3 * 4);
static constexpr size_t OFF_END  = AL(OFF_CNT  + (size_t)CNNODE * 4);

// ---------------- WMMA helpers ----------------
__device__ inline v8f vzero() {
  v8f z = {0.f, 0.f, 0.f, 0.f, 0.f, 0.f, 0.f, 0.f};
  return z;
}

// Load a 16x32 bf16 A-fragment from a row-major LDS tile (ISA 7.12.2 layout):
// lane L<16 holds row L, K {0..7}∪{16..23}; lane L+16 holds row L, K {8..15}∪{24..31}.
__device__ inline v16bf ldfrag(const __bf16* base, int strideElems) {
  const int lane = threadIdx.x & 31;
  const int r = lane & 15, hi = lane >> 4;
  union { v16bf v; uint4 u[2]; } f;
  const __bf16* p = base + (size_t)r * strideElems + hi * 8;
  f.u[0] = *(const uint4*)p;
  f.u[1] = *(const uint4*)(p + 16);
  return f.v;
}

__device__ inline v8f wmma_bf(v16bf a, v16bf b, v8f c) {
  return __builtin_amdgcn_wmma_f32_16x16x32_bf16(false, a, false, b, (short)0, c,
                                                 false, false);
}

// D = A[16 x 32*kc] @ W[32*kc x 32]; W staged transposed (WT[n][k], stride wStride).
__device__ inline void gemm_bf(const __bf16* aLds, int aStride,
                               const __bf16* wT, int wStride, int kc,
                               v8f& d0, v8f& d1) {
  d0 = vzero(); d1 = vzero();
  #pragma unroll 4
  for (int c = 0; c < kc; ++c) {
    v16bf a  = ldfrag(aLds + c * 32, aStride);
    v16bf b0 = ldfrag(wT + c * 32, wStride);
    v16bf b1 = ldfrag(wT + (size_t)16 * wStride + c * 32, wStride);
    d0 = wmma_bf(a, b0, d0);
    d1 = wmma_bf(a, b1, d1);
  }
}

// Store D fragment (C/D layout: VGPR v -> row v+8*hi, col = lane&15) to LDS.
__device__ inline void st_d_bf(__bf16* lds, int stride, int colBase, v8f d) {
  const int lane = threadIdx.x & 31;
  const int n = lane & 15, hi = lane >> 4;
  #pragma unroll
  for (int v = 0; v < 8; ++v)
    lds[(size_t)(v + 8 * hi) * stride + colBase + n] = (__bf16)d[v];
}
__device__ inline void st_d_f(float* lds, int stride, int colBase, v8f d) {
  const int lane = threadIdx.x & 31;
  const int n = lane & 15, hi = lane >> 4;
  #pragma unroll
  for (int v = 0; v < 8; ++v)
    lds[(size_t)(v + 8 * hi) * stride + colBase + n] = d[v];
}

__device__ inline v8f bias_relu(v8f x, float b) {
  #pragma unroll
  for (int i = 0; i < 8; ++i) x[i] = fmaxf(x[i] + b, 0.f);
  return x;
}
__device__ inline v8f bias_silu(v8f x, float b) {
  #pragma unroll
  for (int i = 0; i < 8; ++i) {
    float v = x[i] + b;
    x[i] = v / (1.f + __expf(-v));
  }
  return x;
}
__device__ inline v8f bias_add(v8f x, float b) {
  #pragma unroll
  for (int i = 0; i < 8; ++i) x[i] += b;
  return x;
}

// ---------------- kernel 0: zero accumulators ----------------
__global__ void zero_kernel(float* p, long n) {
  long i = (long)blockIdx.x * blockDim.x + threadIdx.x;
  const long st = (long)gridDim.x * blockDim.x;
  for (; i < n; i += st) p[i] = 0.f;
}

// ---------------- kernel 1: build padded + pooled node arrays ----------------
__global__ void build_nodes_kernel(const float* __restrict__ h,
                                   const float* __restrict__ coords,
                                   __bf16* __restrict__ hh_bf,
                                   float* __restrict__ xxo) {
  const size_t i = (size_t)blockIdx.x * blockDim.x + threadIdx.x; // node*32 + d
  if (i >= (size_t)CNNODE * CD) return;
  const int d = (int)(i & 31);
  const size_t node = i >> 5;
  const int b = (int)(node / CNTOT);
  const int u = (int)(node - (size_t)b * CNTOT);
  float hv;
  if (u < CNPAD) {
    int src = u - CPAD; src = src < 0 ? 0 : (src > CN - 1 ? CN - 1 : src);
    hv = h[((size_t)b * CN + src) * CD + d];
  } else {
    const int p = u - CNPAD;
    hv = 0.f;
    #pragma unroll
    for (int k = 0; k < CKER; ++k) {
      int src = p * CSTR + k - CPAD;
      src = src < 0 ? 0 : (src > CN - 1 ? CN - 1 : src);
      hv += h[((size_t)b * CN + src) * CD + d];
    }
    hv *= (1.f / 3.f);
  }
  hh_bf[node * CD + d] = (__bf16)hv;
  if (d < 3) {
    float xv;
    if (u < CNPAD) {
      int src = u - CPAD; src = src < 0 ? 0 : (src > CN - 1 ? CN - 1 : src);
      xv = coords[((size_t)b * CN + src) * 3 + d];
    } else {
      const int p = u - CNPAD;
      xv = 0.f;
      #pragma unroll
      for (int k = 0; k < CKER; ++k) {
        int src = p * CSTR + k - CPAD;
        src = src < 0 ? 0 : (src > CN - 1 ? CN - 1 : src);
        xv += coords[((size_t)b * CN + src) * 3 + d];
      }
      xv *= (1.f / 3.f);
    }
    xxo[node * 3 + d] = xv;
  }
}

// ---------------- kernel 2: hN0 = hh @ wei + bei (WMMA) ----------------
__global__ __launch_bounds__(128) void embed_kernel(
    const __bf16* __restrict__ hh_bf, const float* __restrict__ wei,
    const float* __restrict__ bei, float* __restrict__ hN0,
    __bf16* __restrict__ hN_bf) {
  __shared__ __align__(16) __bf16 sW[32 * 32];
  __shared__ float sBias[32];
  __shared__ __align__(16) __bf16 sA[4][16 * 32];

  const int t = threadIdx.x;
  for (int i = t; i < 32 * 32; i += blockDim.x) {
    const int n = i >> 5, k = i & 31;
    sW[i] = (__bf16)wei[k * 32 + n];            // transposed stage
  }
  if (t < 32) sBias[t] = bei[t];
  __syncthreads();

  const int wave = t >> 5, lane = t & 31;
  const int r = lane & 15, hi = lane >> 4;
  const long tile = (long)blockIdx.x * 4 + wave;
  const size_t nodeBase = (size_t)tile * 16;
  __bf16* A = sA[wave];

  { // gather 16x32 tile
    const uint4* ps = (const uint4*)(hh_bf + (nodeBase + r) * CD + hi * 16);
    uint4* pd = (uint4*)(A + (size_t)r * 32 + hi * 16);
    pd[0] = ps[0]; pd[1] = ps[1];
  }
  WAVE_LDS_SYNC();

  v8f d0, d1;
  gemm_bf(A, 32, sW, 32, 1, d0, d1);
  const float b0 = sBias[lane & 15], b1 = sBias[16 + (lane & 15)];
  d0 = bias_add(d0, b0);
  d1 = bias_add(d1, b1);

  const int n = lane & 15;
  #pragma unroll
  for (int v = 0; v < 8; ++v) {
    const size_t node = nodeBase + v + 8 * hi;
    hN0[node * CD + n]        = d0[v];
    hN0[node * CD + 16 + n]   = d1[v];
    hN_bf[node * CD + n]      = (__bf16)d0[v];
    hN_bf[node * CD + 16 + n] = (__bf16)d1[v];
  }
}

// ---------------- kernel 3: fused edge pipeline (WMMA) ----------------
__global__ __launch_bounds__(128) void edge_kernel(
    const int* __restrict__ eidx, const __bf16* __restrict__ hh_bf,
    const __bf16* __restrict__ hN_bf, const float* __restrict__ xx,
    const float* __restrict__ em_w1, const float* __restrict__ em_b1,
    const float* __restrict__ em_w2, const float* __restrict__ em_b2,
    const float* __restrict__ em_w3, const float* __restrict__ em_b3,
    const float* __restrict__ lne_g, const float* __restrict__ lne_b,
    const float* __restrict__ ge_w1, const float* __restrict__ ge_b1,
    const float* __restrict__ ge_w2, const float* __restrict__ ge_b2,
    const float* __restrict__ gc_w1, const float* __restrict__ gc_b1,
    const float* __restrict__ gc_w2, const float* __restrict__ gx_w1,
    const float* __restrict__ gx_b1, const float* __restrict__ gx_w2,
    float* __restrict__ magg, float* __restrict__ xacc,
    float* __restrict__ cnt) {
  __shared__ __align__(16) __bf16 sEmW1[32 * 64];
  __shared__ __align__(16) __bf16 sEmW2[32 * 32];
  __shared__ __align__(16) __bf16 sEmW3[32 * 32];
  __shared__ __align__(16) __bf16 sGeW1[32 * 128];
  __shared__ __align__(16) __bf16 sGeW2[32 * 32];
  __shared__ __align__(16) __bf16 sGcW1[32 * 32];
  __shared__ __align__(16) __bf16 sGxW1[32 * 32];
  __shared__ float sB[11 * 32];
  __shared__ __align__(16) __bf16 sTile[4][16 * 128];
  __shared__ __align__(16) __bf16 sTileB[4][16 * 32];
  __shared__ float sF[4][16 * 32];
  __shared__ float sEF[4][16 * 8];
  __shared__ int   sND[4][32];

  const int t = threadIdx.x;
  for (int i = t; i < 32 * 64; i += blockDim.x) {
    const int n = i >> 6, k = i & 63;
    sEmW1[i] = (__bf16)em_w1[k * 32 + n];
  }
  for (int i = t; i < 32 * 32; i += blockDim.x) {
    const int n = i >> 5, k = i & 31;
    sEmW2[i] = (__bf16)em_w2[k * 32 + n];
    sEmW3[i] = (__bf16)em_w3[k * 32 + n];
    sGeW2[i] = (__bf16)ge_w2[k * 32 + n];
    sGcW1[i] = (__bf16)gc_w1[k * 32 + n];
    sGxW1[i] = (__bf16)gx_w1[k * 32 + n];
  }
  for (int i = t; i < 32 * 128; i += blockDim.x) {
    const int n = i >> 7, k = i & 127;
    sGeW1[i] = (k < 97) ? (__bf16)ge_w1[k * 32 + n] : (__bf16)0.f; // zero-pad K 97..127
  }
  if (t < 32) {
    sB[0 * 32 + t] = em_b1[t]; sB[1 * 32 + t] = em_b2[t]; sB[2 * 32 + t] = em_b3[t];
    sB[3 * 32 + t] = lne_g[t]; sB[4 * 32 + t] = lne_b[t];
    sB[5 * 32 + t] = ge_b1[t]; sB[6 * 32 + t] = ge_b2[t];
    sB[7 * 32 + t] = gc_b1[t]; sB[8 * 32 + t] = gx_b1[t];
    sB[9 * 32 + t] = gc_w2[t]; sB[10 * 32 + t] = gx_w2[t];
  }
  __syncthreads();

  const int wave = t >> 5, lane = t & 31;
  const int r = lane & 15, hi = lane >> 4;
  const long tile = (long)blockIdx.x * 4 + wave;  // 16 edges per wave
  __bf16* T  = sTile[wave];
  __bf16* TB = sTileB[wave];
  float*  F  = sF[wave];
  float*  EF = sEF[wave];
  int*    ND = sND[wave];

  __builtin_prefetch(eidx + (size_t)((tile * 16) % ((long)CB * CE)), 0, 0);

  // ---- edge meta: row/col nodes + geometry (lanes 0..15, one edge each) ----
  if (lane < 16) {
    const long j0 = tile * 16 + lane;
    const int b = (int)(j0 / CEPG);
    const int j = (int)(j0 - (long)b * CEPG);
    int rn, cn;
    if (j < CE) {
      rn = eidx[(size_t)b * CE + j] + CPAD;
      cn = eidx[(size_t)CB * CE + (size_t)b * CE + j] + CPAD;
    } else {
      int tt = j - CE;
      if (tt < CPK) {
        const int q = tt / CKER, k = tt - q * CKER;
        rn = CNPAD + q; cn = q * CSTR + k;
      } else {
        tt -= CPK;
        const int q = tt / CKER, k = tt - q * CKER;
        rn = q * CSTR + k; cn = CNPAD + q;
      }
    }
    rn += b * CNTOT; cn += b * CNTOT;
    ND[lane] = rn; ND[16 + lane] = cn;
    const float xr0 = xx[(size_t)rn * 3 + 0], xr1 = xx[(size_t)rn * 3 + 1],
                xr2 = xx[(size_t)rn * 3 + 2];
    const float xc0 = xx[(size_t)cn * 3 + 0], xc1 = xx[(size_t)cn * 3 + 1],
                xc2 = xx[(size_t)cn * 3 + 2];
    const float d0 = xr0 - xc0, d1 = xr1 - xc1, d2 = xr2 - xc2;
    float c0 = xr1 * xc2 - xr2 * xc1;
    float c1 = xr2 * xc0 - xr0 * xc2;
    float c2 = xr0 * xc1 - xr1 * xc0;
    const float nrm = sqrtf(c0 * c0 + c1 * c1 + c2 * c2) + 1.0f;
    c0 /= nrm; c1 /= nrm; c2 /= nrm;
    float* e = EF + lane * 8;
    e[0] = d0; e[1] = d1; e[2] = d2;
    e[3] = c0; e[4] = c1; e[5] = c2;
    e[6] = d0 * d0 + d1 * d1 + d2 * d2; // rad
  }
  WAVE_LDS_SYNC();

  // ---- gather h_row | h_col into tile cols 0..63 (bf16) ----
  {
    const int rn = ND[r], cn = ND[16 + r];
    const uint4* ps = (const uint4*)(hh_bf + (size_t)rn * CD + hi * 16);
    uint4* pd = (uint4*)(T + (size_t)r * 128 + hi * 16);
    pd[0] = ps[0]; pd[1] = ps[1];
    const uint4* qs = (const uint4*)(hh_bf + (size_t)cn * CD + hi * 16);
    uint4* qd = (uint4*)(T + (size_t)r * 128 + 32 + hi * 16);
    qd[0] = qs[0]; qd[1] = qs[1];
  }
  WAVE_LDS_SYNC();

  v8f a0, a1;
  // edge MLP layer 1: [h_row|h_col](16x64) @ em_w1 -> relu, write cols 64..95
  gemm_bf(T, 128, sEmW1, 64, 2, a0, a1);
  a0 = bias_relu(a0, sB[0 * 32 + (lane & 15)]);
  a1 = bias_relu(a1, sB[0 * 32 + 16 + (lane & 15)]);
  st_d_bf(T, 128, 64, a0); st_d_bf(T, 128, 80, a1);
  WAVE_LDS_SYNC();
  // layer 2 -> relu, write cols 0..31
  gemm_bf(T + 64, 128, sEmW2, 32, 1, a0, a1);
  a0 = bias_relu(a0, sB[1 * 32 + (lane & 15)]);
  a1 = bias_relu(a1, sB[1 * 32 + 16 + (lane & 15)]);
  st_d_bf(T, 128, 0, a0); st_d_bf(T, 128, 16, a1);
  WAVE_LDS_SYNC();
  // layer 3 (linear) -> f32 scratch for LayerNorm
  gemm_bf(T, 128, sEmW3, 32, 1, a0, a1);
  a0 = bias_add(a0, sB[2 * 32 + (lane & 15)]);
  a1 = bias_add(a1, sB[2 * 32 + 16 + (lane & 15)]);
  st_d_f(F, 32, 0, a0); st_d_f(F, 32, 16, a1);
  WAVE_LDS_SYNC();

  // ---- LayerNorm(ea) + assemble message-input tail [rad | ea | 0-pad] ----
  if (lane < 16) {
    const float* row = F + lane * 32;
    float mu = 0.f;
    #pragma unroll
    for (int d = 0; d < 32; ++d) mu += row[d];
    mu *= 0.03125f;
    float var = 0.f;
    #pragma unroll
    for (int d = 0; d < 32; ++d) { const float dd = row[d] - mu; var += dd * dd; }
    var *= 0.03125f;
    const float inv = rsqrtf(var + 1e-5f);
    __bf16* tr = T + (size_t)lane * 128;
    tr[64] = (__bf16)EF[lane * 8 + 6];
    #pragma unroll
    for (int d = 0; d < 32; ++d)
      tr[65 + d] = (__bf16)((row[d] - mu) * inv * sB[3 * 32 + d] + sB[4 * 32 + d]);
    for (int d = 97; d < 128; ++d) tr[d] = (__bf16)0.f;
  }
  // gather hN_row | hN_col into cols 0..63
  {
    const int rn = ND[r], cn = ND[16 + r];
    const uint4* ps = (const uint4*)(hN_bf + (size_t)rn * CD + hi * 16);
    uint4* pd = (uint4*)(T + (size_t)r * 128 + hi * 16);
    pd[0] = ps[0]; pd[1] = ps[1];
    const uint4* qs = (const uint4*)(hN_bf + (size_t)cn * CD + hi * 16);
    uint4* qd = (uint4*)(T + (size_t)r * 128 + 32 + hi * 16);
    qd[0] = qs[0]; qd[1] = qs[1];
  }
  WAVE_LDS_SYNC();

  // ---- message MLP: K=128 (zero-padded 97) then 32 ----
  gemm_bf(T, 128, sGeW1, 128, 4, a0, a1);
  a0 = bias_silu(a0, sB[5 * 32 + (lane & 15)]);
  a1 = bias_silu(a1, sB[5 * 32 + 16 + (lane & 15)]);
  st_d_bf(TB, 32, 0, a0); st_d_bf(TB, 32, 16, a1);
  WAVE_LDS_SYNC();
  v8f m0, m1;
  gemm_bf(TB, 32, sGeW2, 32, 1, m0, m1);
  m0 = bias_silu(m0, sB[6 * 32 + (lane & 15)]);
  m1 = bias_silu(m1, sB[6 * 32 + 16 + (lane & 15)]);
  st_d_bf(T, 128, 0, m0); st_d_bf(T, 128, 16, m1); // m as next A operand
  WAVE_LDS_SYNC();

  // ---- phi_x gate: silu(m@gc_w1+b) . gc_w2 ----
  gemm_bf(T, 128, sGcW1, 32, 1, a0, a1);
  a0 = bias_silu(a0, sB[7 * 32 + (lane & 15)]);
  a1 = bias_silu(a1, sB[7 * 32 + 16 + (lane & 15)]);
  st_d_f(F, 32, 0, a0); st_d_f(F, 32, 16, a1);
  WAVE_LDS_SYNC();
  if (lane < 16) {
    float s = 0.f;
    #pragma unroll
    for (int d = 0; d < 32; ++d) s += F[lane * 32 + d] * sB[9 * 32 + d];
    EF[lane * 8 + 7] = s; // phi_x
  }
  WAVE_LDS_SYNC();

  // ---- phi_c gate + coordinate scatter ----
  gemm_bf(T, 128, sGxW1, 32, 1, a0, a1);
  a0 = bias_silu(a0, sB[8 * 32 + (lane & 15)]);
  a1 = bias_silu(a1, sB[8 * 32 + 16 + (lane & 15)]);
  st_d_f(F, 32, 0, a0); st_d_f(F, 32, 16, a1);
  WAVE_LDS_SYNC();
  if (lane < 16) {
    float s = 0.f;
    #pragma unroll
    for (int d = 0; d < 32; ++d) s += F[lane * 32 + d] * sB[10 * 32 + d];
    const float px = EF[lane * 8 + 7];
    const float* e = EF + lane * 8;
    const int rn = ND[lane];
    atomicAdd(&xacc[(size_t)rn * 3 + 0], e[0] * px + e[3] * s);
    atomicAdd(&xacc[(size_t)rn * 3 + 1], e[1] * px + e[4] * s);
    atomicAdd(&xacc[(size_t)rn * 3 + 2], e[2] * px + e[5] * s);
    atomicAdd(&cnt[rn], 1.0f);
  }

  // ---- message scatter: segment_sum(m, row) ----
  {
    const int n = lane & 15;
    #pragma unroll
    for (int v = 0; v < 8; ++v) {
      const int node = ND[v + 8 * hi]; // row node of edge (v + 8*hi)
      atomicAdd(&magg[(size_t)node * CD + n], m0[v]);
      atomicAdd(&magg[(size_t)node * CD + 16 + n], m1[v]);
    }
  }
}

// ---------------- kernel 4: node update + LN + output (pooled only) ----------------
__global__ __launch_bounds__(128) void node_update_kernel(
    const float* __restrict__ hN0, const __bf16* __restrict__ hN_bf,
    const float* __restrict__ magg, const float* __restrict__ xx,
    const float* __restrict__ xacc, const float* __restrict__ cnt,
    const float* __restrict__ gn_w1, const float* __restrict__ gn_b1,
    const float* __restrict__ gn_w2, const float* __restrict__ gn_b2,
    const float* __restrict__ weo, const float* __restrict__ beo,
    const float* __restrict__ lnh_g, const float* __restrict__ lnh_b,
    float* __restrict__ out) {
  __shared__ __align__(16) __bf16 sW1[32 * 64];
  __shared__ __align__(16) __bf16 sW2[32 * 32];
  __shared__ __align__(16) __bf16 sWo[32 * 32];
  __shared__ float sB[5 * 32];
  __shared__ __align__(16) __bf16 sA[4][16 * 64];
  __shared__ __align__(16) __bf16 sT[4][16 * 32];
  __shared__ float sF[4][16 * 32];

  const int t = threadIdx.x;
  for (int i = t; i < 32 * 64; i += blockDim.x) {
    const int n = i >> 6, k = i & 63;
    sW1[i] = (__bf16)gn_w1[k * 32 + n];
  }
  for (int i = t; i < 32 * 32; i += blockDim.x) {
    const int n = i >> 5, k = i & 31;
    sW2[i] = (__bf16)gn_w2[k * 32 + n];
    sWo[i] = (__bf16)weo[k * 32 + n];
  }
  if (t < 32) {
    sB[0 * 32 + t] = gn_b1[t]; sB[1 * 32 + t] = gn_b2[t]; sB[2 * 32 + t] = beo[t];
    sB[3 * 32 + t] = lnh_g[t]; sB[4 * 32 + t] = lnh_b[t];
  }
  __syncthreads();

  const int wave = t >> 5, lane = t & 31;
  const int r = lane & 15, hi = lane >> 4;
  const long tile = (long)blockIdx.x * 4 + wave;
  const long pbase = tile * 16;                 // pooled flat index base
  const int bI = (int)(pbase / CP);
  const int p0 = (int)(pbase - (long)bI * CP);
  const size_t gBase = (size_t)bI * CNTOT + CNPAD + p0; // contiguous 16 nodes
  __bf16* A = sA[wave];
  __bf16* TB = sT[wave];
  float* F = sF[wave];

  // gather [hN0_bf | magg_bf] 16x64
  {
    const size_t node = gBase + r;
    const uint4* ps = (const uint4*)(hN_bf + node * CD + hi * 16);
    uint4* pd = (uint4*)(A + (size_t)r * 64 + hi * 16);
    pd[0] = ps[0]; pd[1] = ps[1];
    #pragma unroll
    for (int d = 0; d < 16; ++d)
      A[(size_t)r * 64 + 32 + hi * 16 + d] = (__bf16)magg[node * CD + hi * 16 + d];
  }
  WAVE_LDS_SYNC();

  v8f u0, u1;
  gemm_bf(A, 64, sW1, 64, 2, u0, u1);
  u0 = bias_silu(u0, sB[0 * 32 + (lane & 15)]);
  u1 = bias_silu(u1, sB[0 * 32 + 16 + (lane & 15)]);
  st_d_bf(TB, 32, 0, u0); st_d_bf(TB, 32, 16, u1);
  WAVE_LDS_SYNC();
  gemm_bf(TB, 32, sW2, 32, 1, u0, u1);
  u0 = bias_add(u0, sB[1 * 32 + (lane & 15)]);
  u1 = bias_add(u1, sB[1 * 32 + 16 + (lane & 15)]);
  { // residual: + hN0 (f32)
    const int n = lane & 15;
    #pragma unroll
    for (int v = 0; v < 8; ++v) {
      const size_t node = gBase + v + 8 * hi;
      u0[v] += hN0[node * CD + n];
      u1[v] += hN0[node * CD + 16 + n];
    }
  }
  st_d_bf(A, 64, 0, u0); st_d_bf(A, 64, 16, u1);
  WAVE_LDS_SYNC();
  gemm_bf(A, 64, sWo, 32, 1, u0, u1);
  u0 = bias_add(u0, sB[2 * 32 + (lane & 15)]);
  u1 = bias_add(u1, sB[2 * 32 + 16 + (lane & 15)]);
  st_d_f(F, 32, 0, u0); st_d_f(F, 32, 16, u1);
  WAVE_LDS_SYNC();

  if (lane < 16) {
    // final LayerNorm -> h_pool output
    const float* row = F + lane * 32;
    float mu = 0.f;
    #pragma unroll
    for (int d = 0; d < 32; ++d) mu += row[d];
    mu *= 0.03125f;
    float var = 0.f;
    #pragma unroll
    for (int d = 0; d < 32; ++d) { const float dd = row[d] - mu; var += dd * dd; }
    var *= 0.03125f;
    const float inv = rsqrtf(var + 1e-5f);
    float* orow = out + (size_t)(pbase + lane) * CD;
    #pragma unroll
    for (int d = 0; d < 32; ++d)
      orow[d] = (row[d] - mu) * inv * sB[3 * 32 + d] + sB[4 * 32 + d];
    // x_pool output
    const size_t g = gBase + lane;
    const float c = fmaxf(cnt[g], 1.0f);
    #pragma unroll
    for (int d = 0; d < 3; ++d)
      out[HOUT + (size_t)(pbase + lane) * 3 + d] =
          xx[g * 3 + d] + xacc[g * 3 + d] / c;
  }
}

// ---------------- host launcher ----------------
extern "C" void kernel_launch(void* const* d_in, const int* in_sizes, int n_in,
                              void* d_out, int out_size, void* d_ws, size_t ws_size,
                              hipStream_t stream) {
  (void)in_sizes; (void)n_in; (void)out_size; (void)ws_size;
  const float* h      = (const float*)d_in[0];
  const float* coords = (const float*)d_in[1];
  const int*   eidx   = (const int*)d_in[2];
  const float* em_w1 = (const float*)d_in[3];
  const float* em_b1 = (const float*)d_in[4];
  const float* em_w2 = (const float*)d_in[5];
  const float* em_b2 = (const float*)d_in[6];
  const float* em_w3 = (const float*)d_in[7];
  const float* em_b3 = (const float*)d_in[8];
  const float* lne_g = (const float*)d_in[9];
  const float* lne_b = (const float*)d_in[10];
  const float* wei   = (const float*)d_in[11];
  const float* bei   = (const float*)d_in[12];
  const float* weo   = (const float*)d_in[13];
  const float* beo   = (const float*)d_in[14];
  const float* ge_w1 = (const float*)d_in[15];
  const float* ge_b1 = (const float*)d_in[16];
  const float* ge_w2 = (const float*)d_in[17];
  const float* ge_b2 = (const float*)d_in[18];
  const float* gn_w1 = (const float*)d_in[19];
  const float* gn_b1 = (const float*)d_in[20];
  const float* gn_w2 = (const float*)d_in[21];
  const float* gn_b2 = (const float*)d_in[22];
  const float* gc_w1 = (const float*)d_in[23];
  const float* gc_b1 = (const float*)d_in[24];
  const float* gc_w2 = (const float*)d_in[25];
  const float* gx_w1 = (const float*)d_in[26];
  const float* gx_b1 = (const float*)d_in[27];
  const float* gx_w2 = (const float*)d_in[28];
  const float* lnh_g = (const float*)d_in[29];
  const float* lnh_b = (const float*)d_in[30];

  char* ws = (char*)d_ws;
  __bf16* hh_bf = (__bf16*)(ws + OFF_HHBF);
  float*  xx    = (float*)(ws + OFF_XX);
  float*  hN0   = (float*)(ws + OFF_HN0);
  __bf16* hN_bf = (__bf16*)(ws + OFF_HNBF);
  float*  magg  = (float*)(ws + OFF_MAGG);
  float*  xacc  = (float*)(ws + OFF_XACC);
  float*  cnt   = (float*)(ws + OFF_CNT);
  float*  out   = (float*)d_out;

  zero_kernel<<<8192, 256, 0, stream>>>((float*)(ws + OFF_MAGG),
                                        (long)((OFF_END - OFF_MAGG) / 4));
  build_nodes_kernel<<<(int)(((size_t)CNNODE * CD) / 256), 256, 0, stream>>>(
      h, coords, hh_bf, xx);
  embed_kernel<<<CNNODE / 16 / 4, 128, 0, stream>>>(hh_bf, wei, bei, hN0, hN_bf);
  edge_kernel<<<(int)(CNEDGE / 16 / 4), 128, 0, stream>>>(
      eidx, hh_bf, hN_bf, xx, em_w1, em_b1, em_w2, em_b2, em_w3, em_b3, lne_g,
      lne_b, ge_w1, ge_b1, ge_w2, ge_b2, gc_w1, gc_b1, gc_w2, gx_w1, gx_b1,
      gx_w2, magg, xacc, cnt);
  node_update_kernel<<<(CB * CP) / 16 / 4, 128, 0, stream>>>(
      hN0, hN_bf, magg, xx, xacc, cnt, gn_w1, gn_b1, gn_w2, gn_b2, weo, beo,
      lnh_g, lnh_b, out);
}
